// CrossModalityAttention_29626684408315
// MI455X (gfx1250) — matrix operand order repaired and checked by reading, not compiled, not run on hardware
//
#include <hip/hip_runtime.h>
#include <hip/hip_bf16.h>

// ---------------------------------------------------------------------------
// CrossModalityAttention for MI455X (gfx1250), bf16 WMMA pipeline + TDM.
//   B=2, C=256, H=W=48, N=2304 tokens/batch.  ~106 GFLOP, compute-bound.
// Pipeline:
//   prep_x    : Xbf = bf16(feat + pos), token-major [3][B*N][C]
//   prep_w    : bf16 copies of q/k/v/gate/out weights [5][256][256]
//   proj_gemm : Q(*scale), K bf16 token-major; V bf16 transposed [b][c][n];
//               gate = sigmoid(.) fp32          (v_wmma_f32_16x16x32_bf16)
//   attn      : flash-attention over 3 modalities' keys; K/V blocks staged to
//               LDS by the Tensor Data Mover (tensor_load_to_lds, TENSORcnt),
//               TDM pad feature produces the bank-conflict-free LDS pitches;
//               online softmax, fp32 ctx accumulators, ctx -> ws (fp32)
//   fuse      : fused = g*ctx + (1-g)*x  -> bf16 token-major
//   out_gemm  : out = fused @ out_w^T + out_b -> d_out fp32 [3][B][C][H*W]
// Workspace: ~64.4 MB (see offsets below).
// ---------------------------------------------------------------------------

typedef __bf16        v16bf __attribute__((ext_vector_type(16)));
typedef float         v8f   __attribute__((ext_vector_type(8)));
typedef int           v8i   __attribute__((ext_vector_type(8)));
typedef unsigned int  u32x4 __attribute__((ext_vector_type(4)));
typedef unsigned int  u32x8 __attribute__((ext_vector_type(8)));

#define DEVINL __device__ __forceinline__

constexpr int   Bb   = 2;
constexpr int   Cc   = 256;
constexpr int   Nt   = 48 * 48;     // 2304 tokens
constexpr int   BN   = Bb * Nt;     // 4608 rows
constexpr float QSCALE = 0.0625f;   // C^-0.5

DEVINL unsigned short f2bf(float f) {
  unsigned int u = __builtin_bit_cast(unsigned int, f);
  u += 0x7FFFu + ((u >> 16) & 1u);  // round-to-nearest-even
  return (unsigned short)(u >> 16);
}

DEVINL v8f zero8() { v8f z = {0.f,0.f,0.f,0.f,0.f,0.f,0.f,0.f}; return z; }

DEVINL v8f bfwmma(v16bf a, v16bf b, v8f c) {
  // D = A(16x32) * B(32x16) + C(16x16), fp32 accumulate
  return __builtin_amdgcn_wmma_f32_16x16x32_bf16(
      false, a, false, b, (short)0, c, false, false);
}

// A-fragment K offset for 16-bit 16x32 A tile (per ISA layout table)
DEVINL int a_k0(int hi, int v) { return hi * 8 + (v < 4 ? 2 * v : 16 + 2 * (v - 4)); }
// B-fragment K offset for 16-bit 32x16 B tile (same K order as A)
DEVINL int b_k0(int hi, int v) { return hi * 8 + (v < 4 ? 2 * v : 16 + 2 * (v - 4)); }

// ---------------------------------------------------------------------------
// Tensor Data Mover: 2D tile Global->LDS, 8-byte elements, pad_amount = 1
// DWORD inserted every (2<<pad_interval) DWORDs (gives padded LDS pitch).
// D# group0: count=1 | lds_addr | 57b global addr | type=2 ("image")
// D# group1: data_size=8B, pad fields, tensor/tile dims + strides (bit-packed)
// Issued by one wave; completion via TENSORcnt.
// ---------------------------------------------------------------------------
DEVINL void tdm_load_2d(unsigned lds_addr, const void* gptr, unsigned pad_interval,
                        unsigned tensor_d0, unsigned tensor_d1,
                        unsigned tile_d0, unsigned tile_d1,
                        unsigned long long stride0_units) {
  unsigned long long ga = (unsigned long long)gptr;
  u32x4 g0;
  g0[0] = 1u;                                                   // count=1 (valid)
  g0[1] = lds_addr;                                             // LDS byte offset
  g0[2] = (unsigned)ga;                                         // global_addr lo
  g0[3] = ((unsigned)(ga >> 32) & 0x01FFFFFFu) | (2u << 30);    // addr hi | type=2
  u32x8 g1;
  g1[0] = (3u << 16) | (1u << 20) | (pad_interval << 22);       // 8B, pad_en, pad_amt=0
  g1[1] = (tensor_d0 & 0xFFFFu) << 16;                          // tensor_dim0[15:0]
  g1[2] = (tensor_d0 >> 16) | ((tensor_d1 & 0xFFFFu) << 16);    // dim0 hi | dim1 lo
  g1[3] = (tensor_d1 >> 16) | (tile_d0 << 16);                  // dim1 hi | tile_dim0
  g1[4] = tile_d1;                                              // tile_dim1 (dim2=0)
  g1[5] = (unsigned)stride0_units;                              // dim0_stride[31:0]
  g1[6] = (unsigned)((stride0_units >> 32) & 0xFFFFu);          // stride hi (stride1=0)
  g1[7] = 0u;
  asm volatile("tensor_load_to_lds %0, %1" :: "s"(g0), "s"(g1) : "memory");
}

// ---------------------------------------------------------------------------
// prep: X = feat + pos  (channel-major read, token-major bf16 scatter write)
// ---------------------------------------------------------------------------
__global__ __launch_bounds__(256) void prep_x_kernel(
    const float* __restrict__ f0, const float* __restrict__ f1,
    const float* __restrict__ f2, const float* __restrict__ pos,
    unsigned short* __restrict__ Xbf) {
  int idx = blockIdx.x * 256 + threadIdx.x;          // over 3*B*C*N
  int i   = idx / (Bb * Cc * Nt);
  int rem = idx - i * (Bb * Cc * Nt);
  int bb  = rem / (Cc * Nt);
  int cn  = rem - bb * (Cc * Nt);
  int c   = cn / Nt;
  int n   = cn - c * Nt;
  const float* f = (i == 0) ? f0 : (i == 1) ? f1 : f2;
  float x = f[rem] + pos[c];
  Xbf[(size_t)i * BN * Cc + (size_t)(bb * Nt + n) * Cc + c] = f2bf(x);
}

__global__ __launch_bounds__(256) void prep_w_kernel(
    const float* __restrict__ qw, const float* __restrict__ kw,
    const float* __restrict__ vw, const float* __restrict__ gw,
    const float* __restrict__ ow, unsigned short* __restrict__ Wbf) {
  int idx = blockIdx.x * 256 + threadIdx.x;          // over 5*65536
  int j = idx >> 16, e = idx & 65535;
  const float* w = (j == 0) ? qw : (j == 1) ? kw : (j == 2) ? vw
                 : (j == 3) ? gw : ow;
  Wbf[idx] = f2bf(w[e]);
}

// ---------------------------------------------------------------------------
// Projection GEMM: Out[r, d] = sum_c X[r,c] * W[d,c] + bias[d]
//   grid (BN/64, C/64, 12), block 128 (4 waves); wave tile = 16 rows x 64 cols
//   z = modality*4 + which   (0=Q scaled bf16, 1=K bf16, 2=V transposed bf16,
//                             3=gate sigmoid fp32)
// ---------------------------------------------------------------------------
__global__ __launch_bounds__(128) void proj_gemm_kernel(
    const unsigned short* __restrict__ Xbf, const unsigned short* __restrict__ Wbf,
    const float* __restrict__ qb, const float* __restrict__ kbias,
    const float* __restrict__ vb, const float* __restrict__ gb,
    unsigned short* __restrict__ Q, unsigned short* __restrict__ K,
    unsigned short* __restrict__ Vt, float* __restrict__ G) {
  const int z = blockIdx.z, i = z >> 2, which = z & 3;
  const int lane = threadIdx.x & 31, wave = threadIdx.x >> 5;
  const int hi = lane >> 4, lo = lane & 15;
  const int rb = blockIdx.x * 64 + wave * 16;
  const int cb = blockIdx.y * 64;
  const unsigned short* A  = Xbf + (size_t)i * BN * Cc;
  const unsigned short* Wm = Wbf + (size_t)which * Cc * Cc;
  const float* bias = (which == 0) ? qb : (which == 1) ? kbias
                    : (which == 2) ? vb : gb;

  v8f acc[4];
#pragma unroll
  for (int t = 0; t < 4; ++t) acc[t] = zero8();

  const unsigned int* Arow = (const unsigned int*)(A + (size_t)(rb + lo) * Cc);
#pragma unroll
  for (int ks = 0; ks < 8; ++ks) {
    v8i ai;
#pragma unroll
    for (int v = 0; v < 8; ++v) ai[v] = (int)Arow[(ks * 32 + a_k0(hi, v)) >> 1];
    v16bf af = __builtin_bit_cast(v16bf, ai);
#pragma unroll
    for (int ct = 0; ct < 4; ++ct) {
      const unsigned int* Wrow =
          (const unsigned int*)(Wm + (size_t)(cb + ct * 16 + lo) * Cc);
      v8i bi;
#pragma unroll
      for (int v = 0; v < 8; ++v) bi[v] = (int)Wrow[(ks * 32 + b_k0(hi, v)) >> 1];
      acc[ct] = bfwmma(af, __builtin_bit_cast(v16bf, bi), acc[ct]);
    }
  }

#pragma unroll
  for (int ct = 0; ct < 4; ++ct) {
    int col = cb + ct * 16 + lo;
    float bv = bias[col];
#pragma unroll
    for (int v = 0; v < 8; ++v) {
      int r = rb + v + 8 * hi;                      // C/D layout: m = v + 8*hi
      float val = acc[ct][v] + bv;
      size_t tok = (size_t)i * BN * Cc + (size_t)r * Cc + col;
      if (which == 0) {
        Q[tok] = f2bf(val * QSCALE);
      } else if (which == 1) {
        K[tok] = f2bf(val);
      } else if (which == 2) {
        int b2 = r / Nt, n = r - b2 * Nt;           // transposed: Vt[b][c][n]
        Vt[((size_t)(i * Bb + b2) * Cc + col) * Nt + n] = f2bf(val);
      } else {
        G[tok] = 1.f / (1.f + __expf(-val));
      }
    }
  }
}

// ---------------------------------------------------------------------------
// Flash attention: grid (Nt/16/4, B, 3), block 128 (4 waves).
// Each wave owns a 16-query tile; WG shares TDM-staged 32-key K / V^T blocks.
//   sK: 32 keys x 256 ch bf16, LDS pitch 129 dwords (TDM pad_interval=6)
//   sV: 256 ch x 32 keys bf16, LDS pitch  17 dwords (TDM pad_interval=3)
// ---------------------------------------------------------------------------
__global__ __launch_bounds__(128) void attn_kernel(
    const unsigned short* __restrict__ Q, const unsigned short* __restrict__ K,
    const unsigned short* __restrict__ Vt, float* __restrict__ CtxOut) {
  __shared__ unsigned int   sK[32 * 129];       // pitch 258 halves
  __shared__ unsigned int   sV[256 * 17];       // pitch 34 halves
  __shared__ unsigned short sP[4][16 * 34];     // per-wave P tile, pitch 34 halves

  const int i = blockIdx.z, bb = blockIdx.y;
  const int tid = threadIdx.x, lane = tid & 31, wave = tid >> 5;
  const int hi = lane >> 4, lo = lane & 15;
  const int qbase = (blockIdx.x * 4 + wave) * 16;

  const unsigned int* Qrow = (const unsigned int*)(
      Q + (size_t)((i * Bb + bb) * Nt + qbase + lo) * Cc);

  v8f acc[16];
#pragma unroll
  for (int t = 0; t < 16; ++t) acc[t] = zero8();
  float mrun[8], lrun[8];
#pragma unroll
  for (int v = 0; v < 8; ++v) { mrun[v] = -1e30f; lrun[v] = 0.f; }

  for (int j = 0; j < 3; ++j) {
    const unsigned short* Kbase = K  + (size_t)(j * Bb + bb) * Nt * Cc;
    const unsigned short* Vbase = Vt + (size_t)(j * Bb + bb) * Cc * Nt;
    for (int kb = 0; kb < Nt / 32; ++kb) {
      __syncthreads();                            // previous block fully consumed
      if (wave == 0) {
        // K block: 32 rows x 512B. 8B units: tile 64x32, row stride 64,
        // pad 1 DWORD per 128 DWORDs -> LDS pitch 129 dwords.
        tdm_load_2d((unsigned)(unsigned long long)(void*)sK,
                    Kbase + (size_t)kb * 32 * Cc,
                    /*pad_interval=*/6u,
                    /*tensor_d0=*/64u, /*tensor_d1=*/32u,
                    /*tile_d0=*/64u,   /*tile_d1=*/32u,
                    /*stride0=*/64ull);
        __builtin_amdgcn_s_wait_tensorcnt(0);
      } else if (wave == 1) {
        // V^T block: 256 rows x 64B. 8B units: tile 8x256, row stride 576,
        // pad 1 DWORD per 16 DWORDs -> LDS pitch 17 dwords.
        tdm_load_2d((unsigned)(unsigned long long)(void*)sV,
                    Vbase + (size_t)kb * 32,
                    /*pad_interval=*/3u,
                    /*tensor_d0=*/576u, /*tensor_d1=*/256u,
                    /*tile_d0=*/8u,     /*tile_d1=*/256u,
                    /*stride0=*/(unsigned long long)(Nt / 4));   // 576
        __builtin_amdgcn_s_wait_tensorcnt(0);
      }
      __syncthreads();

      if (kb + 1 < Nt / 32)                       // -> global_prefetch_b8
        __builtin_prefetch(Kbase + (size_t)(kb + 1) * 32 * Cc + tid * 64, 0, 0);

      // S = Q * K^T   (Q pre-scaled by C^-0.5)
      v8f s0 = zero8(), s1 = zero8();
#pragma unroll
      for (int ks = 0; ks < 8; ++ks) {
        v8i ai;
#pragma unroll
        for (int v = 0; v < 8; ++v) ai[v] = (int)Qrow[(ks * 32 + a_k0(hi, v)) >> 1];
        v16bf af = __builtin_bit_cast(v16bf, ai);
        v8i b0, b1;
#pragma unroll
        for (int v = 0; v < 8; ++v) {
          int kk = (ks * 32 + b_k0(hi, v)) >> 1;
          b0[v] = (int)sK[lo * 129 + kk];
          b1[v] = (int)sK[(16 + lo) * 129 + kk];
        }
        s0 = bfwmma(af, __builtin_bit_cast(v16bf, b0), s0);
        s1 = bfwmma(af, __builtin_bit_cast(v16bf, b1), s1);
      }

      // online softmax: rows live across 16 lanes at fixed VGPR v
      float al[8];
#pragma unroll
      for (int v = 0; v < 8; ++v) {
        float t = fmaxf(s0[v], s1[v]);
        t = fmaxf(t, __shfl_xor(t, 1));
        t = fmaxf(t, __shfl_xor(t, 2));
        t = fmaxf(t, __shfl_xor(t, 4));
        t = fmaxf(t, __shfl_xor(t, 8));
        float mnew = fmaxf(mrun[v], t);
        al[v] = __expf(mrun[v] - mnew);
        mrun[v] = mnew;
        float p0 = __expf(s0[v] - mnew), p1 = __expf(s1[v] - mnew);
        s0[v] = p0; s1[v] = p1;
        float rs = p0 + p1;
        rs += __shfl_xor(rs, 1);
        rs += __shfl_xor(rs, 2);
        rs += __shfl_xor(rs, 4);
        rs += __shfl_xor(rs, 8);
        lrun[v] = lrun[v] * al[v] + rs;
      }
#pragma unroll
      for (int ct = 0; ct < 16; ++ct)
#pragma unroll
        for (int v = 0; v < 8; ++v) acc[ct][v] = acc[ct][v] * al[v];

      // C/D-layout -> A-fragment via per-wave LDS round-trip (in-order DS)
#pragma unroll
      for (int v = 0; v < 8; ++v) {
        int row = v + 8 * hi;
        sP[wave][row * 34 + lo]      = f2bf(s0[v]);
        sP[wave][row * 34 + 16 + lo] = f2bf(s1[v]);
      }
      const unsigned int* Pd = (const unsigned int*)&sP[wave][0];
      v8i pi;
#pragma unroll
      for (int v = 0; v < 8; ++v) pi[v] = (int)Pd[lo * 17 + (a_k0(hi, v) >> 1)];
      v16bf pf = __builtin_bit_cast(v16bf, pi);

      // ctx += P(16x32) * V(32x16) for all 16 channel tiles
#pragma unroll
      for (int ct = 0; ct < 16; ++ct) {
        v8i bi;
#pragma unroll
        for (int v = 0; v < 8; ++v)
          bi[v] = (int)sV[(ct * 16 + lo) * 17 + (b_k0(hi, v) >> 1)];
        acc[ct] = bfwmma(pf, __builtin_bit_cast(v16bf, bi), acc[ct]);
      }
    }
  }

  float inv[8];
#pragma unroll
  for (int v = 0; v < 8; ++v) inv[v] = 1.f / lrun[v];
  float* Cb = CtxOut + (size_t)i * BN * Cc + (size_t)((size_t)bb * Nt + qbase) * Cc;
#pragma unroll
  for (int ct = 0; ct < 16; ++ct)
#pragma unroll
    for (int v = 0; v < 8; ++v)
      Cb[(size_t)(v + 8 * hi) * Cc + ct * 16 + lo] = acc[ct][v] * inv[v];
}

// ---------------------------------------------------------------------------
// fused = gate*ctx + (1-gate)*(feat+pos)  -> bf16 token-major
// ---------------------------------------------------------------------------
__global__ __launch_bounds__(256) void fuse_kernel(
    const float* __restrict__ f0, const float* __restrict__ f1,
    const float* __restrict__ f2, const float* __restrict__ pos,
    const float* __restrict__ G, const float* __restrict__ Ctx,
    unsigned short* __restrict__ Fused) {
  int idx = blockIdx.x * 256 + threadIdx.x;          // over 3*B*C*N
  int i   = idx / (Bb * Cc * Nt);
  int rem = idx - i * (Bb * Cc * Nt);
  int bb  = rem / (Cc * Nt);
  int cn  = rem - bb * (Cc * Nt);
  int c   = cn / Nt;
  int n   = cn - c * Nt;
  const float* f = (i == 0) ? f0 : (i == 1) ? f1 : f2;
  float x = f[rem] + pos[c];
  size_t tok = (size_t)i * BN * Cc + (size_t)(bb * Nt + n) * Cc + c;
  float g = G[tok];
  Fused[tok] = f2bf(g * Ctx[tok] + (1.f - g) * x);
}

// ---------------------------------------------------------------------------
// Output GEMM: out[i][b][d][n] = sum_c fused[i][b*N+n][c]*out_w[d][c]+out_b[d]
//   grid (BN/64, C/64, 3), block 128; stores fp32 channel-major to d_out.
// ---------------------------------------------------------------------------
__global__ __launch_bounds__(128) void out_gemm_kernel(
    const unsigned short* __restrict__ Fused, const unsigned short* __restrict__ Wbf,
    const float* __restrict__ ob, float* __restrict__ Out) {
  const int i = blockIdx.z;
  const int lane = threadIdx.x & 31, wave = threadIdx.x >> 5;
  const int hi = lane >> 4, lo = lane & 15;
  const int rb = blockIdx.x * 64 + wave * 16;
  const int cb = blockIdx.y * 64;
  const unsigned short* A  = Fused + (size_t)i * BN * Cc;
  const unsigned short* Wm = Wbf + (size_t)4 * Cc * Cc;   // out_w

  v8f acc[4];
#pragma unroll
  for (int t = 0; t < 4; ++t) acc[t] = zero8();

  const unsigned int* Arow = (const unsigned int*)(A + (size_t)(rb + lo) * Cc);
#pragma unroll
  for (int ks = 0; ks < 8; ++ks) {
    v8i ai;
#pragma unroll
    for (int v = 0; v < 8; ++v) ai[v] = (int)Arow[(ks * 32 + a_k0(hi, v)) >> 1];
    v16bf af = __builtin_bit_cast(v16bf, ai);
#pragma unroll
    for (int ct = 0; ct < 4; ++ct) {
      const unsigned int* Wrow =
          (const unsigned int*)(Wm + (size_t)(cb + ct * 16 + lo) * Cc);
      v8i bi;
#pragma unroll
      for (int v = 0; v < 8; ++v) bi[v] = (int)Wrow[(ks * 32 + b_k0(hi, v)) >> 1];
      acc[ct] = bfwmma(af, __builtin_bit_cast(v16bf, bi), acc[ct]);
    }
  }

#pragma unroll
  for (int ct = 0; ct < 4; ++ct) {
    int col = cb + ct * 16 + lo;
    float bv = ob[col];
#pragma unroll
    for (int v = 0; v < 8; ++v) {
      int r  = rb + v + 8 * hi;
      int b2 = r / Nt, n = r - b2 * Nt;
      Out[((size_t)(i * Bb + b2) * Cc + col) * Nt + n] = acc[ct][v] + bv;
    }
  }
}

// ---------------------------------------------------------------------------
extern "C" void kernel_launch(void* const* d_in, const int* in_sizes, int n_in,
                              void* d_out, int out_size, void* d_ws, size_t ws_size,
                              hipStream_t stream) {
  const float* feat0 = (const float*)d_in[0];
  const float* feat1 = (const float*)d_in[1];
  const float* feat2 = (const float*)d_in[2];
  const float* pos   = (const float*)d_in[3];
  const float* q_w   = (const float*)d_in[4];
  const float* q_b   = (const float*)d_in[5];
  const float* k_w   = (const float*)d_in[6];
  const float* k_b   = (const float*)d_in[7];
  const float* v_w   = (const float*)d_in[8];
  const float* v_b   = (const float*)d_in[9];
  const float* g_w   = (const float*)d_in[10];
  const float* g_b   = (const float*)d_in[11];
  const float* o_w   = (const float*)d_in[12];
  const float* o_b   = (const float*)d_in[13];
  float* out = (float*)d_out;

  // workspace layout (bytes); total ~64.4 MB
  char* ws = (char*)d_ws;
  constexpr size_t SZ_TOK_BF  = (size_t)3 * BN * Cc * 2;   //  7,077,888
  constexpr size_t SZ_W_BF    = (size_t)5 * Cc * Cc * 2;   //    655,360
  constexpr size_t SZ_TOK_F32 = (size_t)3 * BN * Cc * 4;   // 14,155,776
  unsigned short* Xbf   = (unsigned short*)(ws);
  unsigned short* Wbf   = (unsigned short*)(ws + SZ_TOK_BF);
  unsigned short* Qbf   = (unsigned short*)(ws + SZ_TOK_BF + SZ_W_BF);
  unsigned short* Kbf   = (unsigned short*)(ws + 2 * SZ_TOK_BF + SZ_W_BF);
  unsigned short* Vtbf  = (unsigned short*)(ws + 3 * SZ_TOK_BF + SZ_W_BF);
  float*          Gate  = (float*)(ws + 4 * SZ_TOK_BF + SZ_W_BF);
  float*          Ctx   = (float*)(ws + 4 * SZ_TOK_BF + SZ_W_BF + SZ_TOK_F32);
  unsigned short* Fused = (unsigned short*)(ws + 4 * SZ_TOK_BF + SZ_W_BF + 2 * SZ_TOK_F32);

  const int elem3 = 3 * Bb * Cc * Nt;                      // 3,538,944 (÷256 exact)

  prep_w_kernel<<<(5 * Cc * Cc) / 256, 256, 0, stream>>>(q_w, k_w, v_w, g_w, o_w, Wbf);
  prep_x_kernel<<<elem3 / 256, 256, 0, stream>>>(feat0, feat1, feat2, pos, Xbf);

  proj_gemm_kernel<<<dim3(BN / 64, Cc / 64, 12), 128, 0, stream>>>(
      Xbf, Wbf, q_b, k_b, v_b, g_b, Qbf, Kbf, Vtbf, Gate);

  attn_kernel<<<dim3(Nt / 16 / 4, Bb, 3), 128, 0, stream>>>(Qbf, Kbf, Vtbf, Ctx);

  fuse_kernel<<<elem3 / 256, 256, 0, stream>>>(feat0, feat1, feat2, pos, Gate, Ctx, Fused);

  out_gemm_kernel<<<dim3(BN / 64, Cc / 64, 3), 128, 0, stream>>>(Fused, Wbf, o_b, out);
}
